// SemanticChunker_6640019440250
// MI455X (gfx1250) — matrix operand order repaired and checked by reading, not compiled, not run on hardware
//
#include <hip/hip_runtime.h>
#include <cmath>

#define B_DIM 8
#define S_DIM 4096
#define D_DIM 1024
#define H_DIM 256
#define MAXC 64
#define MAXL 64
#define MINL 4

// LDS B-tile row stride: 256 data + 8 pad bf16 -> 528 bytes (16B aligned,
// avoids the 64-bank degenerate pattern of a 512B stride).
#define BSTRIDE 264

typedef __attribute__((ext_vector_type(16))) __bf16 v16bf;
typedef __attribute__((ext_vector_type(8)))  float  v8f;

// ---------------------------------------------------------------------------
// Kernel 1: W1 fp32 -> bf16 (1024x256 = 256K elements, one-shot prep)
// ---------------------------------------------------------------------------
__global__ void cvt_w1_bf16(const float* __restrict__ w1,
                            __bf16* __restrict__ w1bf, int n) {
    int i = blockIdx.x * blockDim.x + threadIdx.x;
    if (i < n) w1bf[i] = (__bf16)w1[i];
}

// ---------------------------------------------------------------------------
// Kernel 2: fused  boundaries = (GELU(hidden@W1 + b1) @ W2 + b2) > 0
// 4 waves / block, 16 token-rows per wave. Per K-step the 32x256 bf16 W1
// tile (16KB) is staged once into LDS (double-buffered, one barrier/step)
// and shared by all 4 waves. B fragments are software-pipelined depth-2 so
// the DS loads overlap the WMMA issue instead of full dscnt==0 stalls.
// ---------------------------------------------------------------------------
__global__ __launch_bounds__(128)
void boundary_kernel(const float* __restrict__ hidden,
                     const __bf16* __restrict__ w1bf,
                     const float* __restrict__ b1,
                     const float* __restrict__ w2,
                     const float* __restrict__ b2,
                     float* __restrict__ boundaries) {
    __shared__ __bf16 Btile[2][32 * BSTRIDE];   // 2 x 16.5KB

    const int tid      = threadIdx.x;
    const int lane     = tid & 31;
    const int wave     = tid >> 5;
    const int row_base = (blockIdx.x * 4 + wave) * 16;
    const int r        = lane & 15;   // C-layout: N = r ; A-layout: M row = r
    const int g        = lane >> 4;   // half-select

    // Cooperative stage of one 32xH_DIM bf16 K-tile into LDS buffer `buf`.
    auto stage = [&](int buf, int kb) {
        const int row = tid >> 2;                 // 0..31  (K row)
        const int seg = tid & 3;                  // 0..3   (64 bf16 each)
        const uint4* src =
            (const uint4*)(w1bf + (size_t)(kb + row) * H_DIM + seg * 64);
        uint4* dst = (uint4*)(&Btile[buf][row * BSTRIDE + seg * 64]);
        #pragma unroll
        for (int i = 0; i < 8; ++i) dst[i] = src[i];
    };

    v8f acc[16] = {};
    const float* arow_p = hidden + (size_t)(row_base + r) * D_DIM;

    stage(0, 0);
    int t = 0;
    for (int kb = 0; kb < D_DIM; kb += 32, t ^= 1) {
        __syncthreads();                          // tile t ready, t^1 free
        if (kb + 32 < D_DIM) stage(t ^ 1, kb + 32);

        // --- A fragment: 16-bit A 16x32 layout.
        // lane (r,g): elems 0..7  -> K = kb + 8*g + 0..7
        //             elems 8..15 -> K = kb + 16 + 8*g + 0..7
        const float* ap = arow_p + kb + 8 * g;
        float4 x0 = *(const float4*)(ap);
        float4 x1 = *(const float4*)(ap + 4);
        float4 x2 = *(const float4*)(ap + 16);
        float4 x3 = *(const float4*)(ap + 20);
        v16bf a;
        a[0]  = (__bf16)x0.x;  a[1]  = (__bf16)x0.y;
        a[2]  = (__bf16)x0.z;  a[3]  = (__bf16)x0.w;
        a[4]  = (__bf16)x1.x;  a[5]  = (__bf16)x1.y;
        a[6]  = (__bf16)x1.z;  a[7]  = (__bf16)x1.w;
        a[8]  = (__bf16)x2.x;  a[9]  = (__bf16)x2.y;
        a[10] = (__bf16)x2.z;  a[11] = (__bf16)x2.w;
        a[12] = (__bf16)x3.x;  a[13] = (__bf16)x3.y;
        a[14] = (__bf16)x3.z;  a[15] = (__bf16)x3.w;

        // --- B fragments from LDS: lane holds B-col N = 16n+r; elems 0..7 ->
        // K = 8*g + 0..7, elems 8..15 -> K = 16 + 8*g + 0..7.
        // Depth-2 software pipeline so DS latency overlaps WMMA issue.
        const __bf16* bl = &Btile[t][r];
        auto ldb = [&](int nn) {
            v16bf v;
            #pragma unroll
            for (int s = 0; s < 16; ++s) {
                int k = (s < 8) ? (8 * g + s) : (16 + 8 * g + (s - 8));
                v[s] = bl[k * BSTRIDE + nn * 16];
            }
            return v;
        };
        v16bf bcur  = ldb(0);
        v16bf bnext = ldb(1);
        #pragma unroll
        for (int n = 0; n < 16; ++n) {
            acc[n] = __builtin_amdgcn_wmma_f32_16x16x32_bf16(
                false, a, false, bcur, (short)0, acc[n], false, false);
            bcur = bnext;
            if (n + 2 < 16)
                bnext = ldb(n + 2);
        }
    }

    // Epilogue: per C element (M = i + 8g, N = 16n + r):
    //   partial_M += GELU(acc + b1[N]) * W2[N];  reduce over N (16 lanes).
    float partial[8] = {0.f, 0.f, 0.f, 0.f, 0.f, 0.f, 0.f, 0.f};
    const float b2v = b2[0];
    #pragma unroll
    for (int n = 0; n < 16; ++n) {
        float b1v = b1[n * 16 + r];
        float w2v = w2[n * 16 + r];
        #pragma unroll
        for (int i = 0; i < 8; ++i) {
            float x  = acc[n][i] + b1v;
            float gl = 0.5f * x * (1.0f + erff(x * 0.70710678118654752f));
            partial[i] += gl * w2v;
        }
    }
    #pragma unroll
    for (int i = 0; i < 8; ++i) {
        float p = partial[i];
        p += __shfl_xor(p, 1, 32);
        p += __shfl_xor(p, 2, 32);
        p += __shfl_xor(p, 4, 32);
        p += __shfl_xor(p, 8, 32);   // reduction stays inside each 16-lane half
        partial[i] = p + b2v;
    }
    if (r == 0) {
        #pragma unroll
        for (int i = 0; i < 8; ++i) {
            // sigmoid(x) > 0.5  <=>  x > 0
            boundaries[row_base + g * 8 + i] = (partial[i] > 0.0f) ? 1.0f : 0.0f;
        }
    }
}

// ---------------------------------------------------------------------------
// Kernel 3: sequential chunk scan, one wave per batch. Lanes fetch 32
// boundary flags at a time (ballot) so the serial core never stalls on loads.
// ---------------------------------------------------------------------------
__global__ __launch_bounds__(32)
void scan_kernel(const float* __restrict__ boundaries,
                 const float* __restrict__ amask,
                 int* __restrict__ ws_start, int* __restrict__ ws_len,
                 int* __restrict__ ws_valid,
                 float* __restrict__ out_cm, float* __restrict__ out_tm,
                 float* __restrict__ out_nc) {
    const int b    = blockIdx.x;
    const int lane = threadIdx.x;

    // valid_len = sum(attention_mask[b])
    float vs = 0.f;
    for (int t = lane; t < S_DIM; t += 32) vs += amask[b * S_DIM + t];
    vs += __shfl_xor(vs, 1, 32);
    vs += __shfl_xor(vs, 2, 32);
    vs += __shfl_xor(vs, 4, 32);
    vs += __shfl_xor(vs, 8, 32);
    vs += __shfl_xor(vs, 16, 32);
    const int valid_len = (int)vs;

    int cs[MAXC], cl[MAXC], cv[MAXC];       // uniform across lanes
    for (int c = 0; c < MAXC; ++c) { cs[c] = 0; cl[c] = 0; cv[c] = 0; }

    int start = 0, cidx = 0;
    for (int t0 = 0; t0 < S_DIM; t0 += 32) {
        bool bb = boundaries[b * S_DIM + t0 + lane] > 0.5f;
        unsigned m = (unsigned)__ballot(bb);
        for (int u = 0; u < 32; ++u) {      // uniform serial core
            int  tt  = t0 + u;
            bool ib  = ((m >> u) & 1u) || (tt == valid_len - 1);
            int  end = tt + 1;
            int  len = end - start;
            bool accept = ib && ((len >= MINL) || (cidx == 0)) && (cidx < MAXC);
            if (accept) {
                cs[cidx] = start;
                cl[cidx] = (len < MAXL) ? len : MAXL;
                cv[cidx] = 1;
                ++cidx;
            }
            if (ib) start = end;            // start advances on every boundary
        }
    }
    const int nc = (cidx < 1) ? 1 : cidx;

    if (lane == 0) {
        out_nc[b] = (float)nc;
        for (int c = 0; c < MAXC; ++c) {
            ws_start[b * MAXC + c] = cs[c];
            ws_len  [b * MAXC + c] = cl[c];
            ws_valid[b * MAXC + c] = cv[c];
        }
    }
    // masks, lanes cooperative
    for (int c = lane; c < MAXC; c += 32) {
        out_cm[b * MAXC + c] = cv[c] ? 1.f : 0.f;
        for (int j = 0; j < MAXL; ++j)
            out_tm[(b * MAXC + c) * MAXL + j] =
                (cv[c] && j < cl[c]) ? 1.f : 0.f;
    }
}

// ---------------------------------------------------------------------------
// Kernel 4: gather chunks[b][c][j][:] = hidden[b][cstart+j][:] * mask
// One 256-thread block per output row (1024 floats = 256 float4). Streaming.
// ---------------------------------------------------------------------------
__global__ __launch_bounds__(256)
void gather_kernel(const float* __restrict__ hidden,
                   const int* __restrict__ ws_start,
                   const int* __restrict__ ws_len,
                   const int* __restrict__ ws_valid,
                   float* __restrict__ out_chunks) {
    const int blk = blockIdx.x;             // b*4096 + c*64 + j
    const int j   = blk & 63;
    const int c   = (blk >> 6) & 63;
    const int b   = blk >> 12;
    const int rec = b * MAXC + c;

    const int st = ws_start[rec];
    const int ln = ws_len[rec];
    const int v  = ws_valid[rec];
    const bool active = (v != 0) && (j < ln);

    float4 val = make_float4(0.f, 0.f, 0.f, 0.f);
    if (active) {
        int idx = st + j;
        idx = idx < 0 ? 0 : (idx > S_DIM - 1 ? S_DIM - 1 : idx);
        const float4* src =
            (const float4*)(hidden + ((size_t)b * S_DIM + idx) * D_DIM);
        val = src[threadIdx.x];
    }
    float4* dst = (float4*)(out_chunks + (size_t)blk * D_DIM);
    dst[threadIdx.x] = val;
}

// ---------------------------------------------------------------------------
extern "C" void kernel_launch(void* const* d_in, const int* in_sizes, int n_in,
                              void* d_out, int out_size, void* d_ws, size_t ws_size,
                              hipStream_t stream) {
    const float* hidden = (const float*)d_in[0];
    const float* amask  = (const float*)d_in[1];
    const float* W1     = (const float*)d_in[2];
    const float* b1     = (const float*)d_in[3];
    const float* W2     = (const float*)d_in[4];
    const float* b2     = (const float*)d_in[5];
    float* out = (float*)d_out;

    // output layout (flat, return order)
    const size_t N_CHUNKS = (size_t)B_DIM * MAXC * MAXL * D_DIM;  // 33,554,432
    float* out_chunks = out;
    float* out_cm     = out_chunks + N_CHUNKS;
    float* out_tm     = out_cm + (size_t)B_DIM * MAXC;
    float* out_bd     = out_tm + (size_t)B_DIM * MAXC * MAXL;
    float* out_nc     = out_bd + (size_t)B_DIM * S_DIM;

    // workspace layout
    __bf16* w1bf   = (__bf16*)d_ws;                               // 512 KB
    int* ws_start  = (int*)((char*)d_ws + (size_t)D_DIM * H_DIM * sizeof(__bf16));
    int* ws_len    = ws_start + B_DIM * MAXC;
    int* ws_valid  = ws_len   + B_DIM * MAXC;

    const int nW1 = D_DIM * H_DIM;
    cvt_w1_bf16<<<(nW1 + 255) / 256, 256, 0, stream>>>(W1, w1bf, nW1);

    // 32768 token rows / 16 rows-per-wave / 4 waves-per-block = 512 blocks
    boundary_kernel<<<(B_DIM * S_DIM) / 64, 128, 0, stream>>>(
        hidden, w1bf, b1, W2, b2, out_bd);

    scan_kernel<<<B_DIM, 32, 0, stream>>>(
        out_bd, amask, ws_start, ws_len, ws_valid, out_cm, out_tm, out_nc);

    gather_kernel<<<B_DIM * MAXC * MAXL, 256, 0, stream>>>(
        hidden, ws_start, ws_len, ws_valid, out_chunks);
}
